// VQCodebookManager_76544907149912
// MI455X (gfx1250) — matrix operand order repaired and checked by reading, not compiled, not run on hardware
//
#include <hip/hip_runtime.h>
#include <math.h>
#include <stdint.h>

typedef __attribute__((ext_vector_type(16))) __bf16 v16bf;
typedef __attribute__((ext_vector_type(8)))  float  v8f;

#define NUM_Q   1024
#define VEC_DIM 256
#define N_ROWS  (32 * 4096)

#define CT_PER_CHUNK 4                               // code-tiles per LDS chunk
#define CHUNK_BYTES  (CT_PER_CHUNK * 8 * 32 * 32)    // 32 KB
#define N_CHUNKS     (64 / CT_PER_CHUNK)             // 16

// ---------------------------------------------------------------------------
// Prep: W (fp32 [1024 x 256]) -> bf16 in WMMA B-fragment order.
// gid = ((c*8 + t)*32 + lane); element j = W[c*16+(lane&15)][t*32+(lane>>4)*16+j]
// ---------------------------------------------------------------------------
__global__ void prep_wfrag(const float* __restrict__ W, __bf16* __restrict__ Wf) {
  int gid  = blockIdx.x * blockDim.x + threadIdx.x;   // 16384 total
  int lane = gid & 31;
  int t    = (gid >> 5) & 7;
  int c    = gid >> 8;
  int n    = c * 16 + (lane & 15);
  int kb   = t * 32 + ((lane >> 4) * 16);
  const float* src = W + (size_t)n * VEC_DIM + kb;
  v16bf v;
#pragma unroll
  for (int j = 0; j < 16; ++j) v[j] = (__bf16)src[j];
  *(v16bf*)(Wf + (size_t)gid * 16) = v;
}

__global__ void prep_w2(const float* __restrict__ W, float* __restrict__ w2) {
  int k = blockIdx.x * blockDim.x + threadIdx.x;      // 1024
  const float4* row = (const float4*)(W + (size_t)k * VEC_DIM);
  float s = 0.f;
#pragma unroll 4
  for (int i = 0; i < VEC_DIM / 4; ++i) {
    float4 v = row[i];
    s += v.x * v.x + v.y * v.y + v.z * v.z + v.w * v.w;
  }
  w2[k] = s;
}

// ---------------------------------------------------------------------------
// CDNA5 async global->LDS copy helpers (CDNA5 ISA).
// LDS byte address = low 32 bits of generic shared pointer (flat->LDS mapping
// uses addr[31:0]).
// ---------------------------------------------------------------------------
__device__ __forceinline__ void async_b128_to_lds(unsigned lds_off,
                                                  unsigned long long gaddr) {
  asm volatile("global_load_async_to_lds_b128 %0, %1, off"
               :: "v"(lds_off), "v"(gaddr) : "memory");
}
__device__ __forceinline__ void wait_asynccnt0() {
  asm volatile("s_wait_asynccnt 0" ::: "memory");
}

// ---------------------------------------------------------------------------
// Main: one wave handles TWO 16-row tiles of X (B-fragment reuse x2).
// B fragments double-buffered in LDS, staged with async-to-LDS copies shared
// by all 8 waves of the block. Scores via bf16 WMMA, f32 accumulation;
// argmin_k (w2[k] - 2*S); gather fp32 codebook rows into q.
// ---------------------------------------------------------------------------
__global__ void __launch_bounds__(256)
vq_assign_gather(const float* __restrict__ X, const float* __restrict__ W,
                 const __bf16* __restrict__ Wf, const float* __restrict__ w2,
                 float* __restrict__ q) {
  extern __shared__ char smem[];               // 2*32KB B buffers + 1KB indices
  int* s_idx = (int*)(smem + 2 * CHUNK_BYTES); // [8 waves][32 rows]

  const int tid  = threadIdx.x;
  const int lane = tid & 31;
  const int wave = tid >> 5;
  const int mlo  = lane & 15;
  const int hi   = lane >> 4;

  const size_t mBase0 = ((size_t)blockIdx.x * 16 + wave * 2) * 16;
  const size_t mBase1 = mBase0 + 16;

  // A fragments for both row tiles (contiguous k order, same as the Wf B fragments):
  // lane l: row m = l&15, k = (l>>4)*16 + {0..15} (elems 0-15).
  v16bf a0[8], a1[8];
  {
    const float* x0 = X + (mBase0 + (size_t)mlo) * VEC_DIM;
    const float* x1 = X + (mBase1 + (size_t)mlo) * VEC_DIM;
#pragma unroll
    for (int t = 0; t < 8; ++t) {
      const int kb = t * 32 + hi * 16;
      {
        const float4* p0 = (const float4*)(x0 + kb);
        const float4* p1 = (const float4*)(x0 + kb + 8);
        float4 f0 = p0[0], f1 = p0[1], f2 = p1[0], f3 = p1[1];
        a0[t][0]=(__bf16)f0.x; a0[t][1]=(__bf16)f0.y; a0[t][2]=(__bf16)f0.z; a0[t][3]=(__bf16)f0.w;
        a0[t][4]=(__bf16)f1.x; a0[t][5]=(__bf16)f1.y; a0[t][6]=(__bf16)f1.z; a0[t][7]=(__bf16)f1.w;
        a0[t][8]=(__bf16)f2.x; a0[t][9]=(__bf16)f2.y; a0[t][10]=(__bf16)f2.z; a0[t][11]=(__bf16)f2.w;
        a0[t][12]=(__bf16)f3.x; a0[t][13]=(__bf16)f3.y; a0[t][14]=(__bf16)f3.z; a0[t][15]=(__bf16)f3.w;
      }
      {
        const float4* p0 = (const float4*)(x1 + kb);
        const float4* p1 = (const float4*)(x1 + kb + 8);
        float4 f0 = p0[0], f1 = p0[1], f2 = p1[0], f3 = p1[1];
        a1[t][0]=(__bf16)f0.x; a1[t][1]=(__bf16)f0.y; a1[t][2]=(__bf16)f0.z; a1[t][3]=(__bf16)f0.w;
        a1[t][4]=(__bf16)f1.x; a1[t][5]=(__bf16)f1.y; a1[t][6]=(__bf16)f1.z; a1[t][7]=(__bf16)f1.w;
        a1[t][8]=(__bf16)f2.x; a1[t][9]=(__bf16)f2.y; a1[t][10]=(__bf16)f2.z; a1[t][11]=(__bf16)f2.w;
        a1[t][12]=(__bf16)f3.x; a1[t][13]=(__bf16)f3.y; a1[t][14]=(__bf16)f3.z; a1[t][15]=(__bf16)f3.w;
      }
    }
  }

  float bestVal0[8], bestVal1[8];
  int   bestIdx0[8], bestIdx1[8];
#pragma unroll
  for (int r = 0; r < 8; ++r) {
    bestVal0[r] = __builtin_inff(); bestIdx0[r] = 0;
    bestVal1[r] = __builtin_inff(); bestIdx1[r] = 0;
  }

  // Stage chunk 0 (32 KB: 256 threads x 128 B each, 8 async b128 per thread).
  {
    const char* src = (const char*)Wf + (size_t)tid * 128;
    unsigned dst = (unsigned)(uintptr_t)(smem) + tid * 128;
#pragma unroll
    for (int i = 0; i < 8; ++i)
      async_b128_to_lds(dst + i * 16, (unsigned long long)(uintptr_t)(src + i * 16));
  }
  wait_asynccnt0();
  __syncthreads();

  for (int ch = 0; ch < N_CHUNKS; ++ch) {
    // Kick off async staging of the next chunk into the other buffer;
    // it overlaps with this chunk's 64 WMMAs.
    if (ch + 1 < N_CHUNKS) {
      const char* src = (const char*)Wf + (size_t)(ch + 1) * CHUNK_BYTES + (size_t)tid * 128;
      unsigned dst = (unsigned)(uintptr_t)(smem) + ((ch + 1) & 1) * CHUNK_BYTES + tid * 128;
#pragma unroll
      for (int i = 0; i < 8; ++i)
        async_b128_to_lds(dst + i * 16, (unsigned long long)(uintptr_t)(src + i * 16));
    }

    const v16bf* bbuf = (const v16bf*)(smem + (ch & 1) * CHUNK_BYTES);
#pragma unroll
    for (int cl = 0; cl < CT_PER_CHUNK; ++cl) {
      const int c = ch * CT_PER_CHUNK + cl;
      v8f acc0 = {}, acc1 = {};
#pragma unroll
      for (int t = 0; t < 8; ++t) {
        v16bf b = bbuf[(cl * 8 + t) * 32 + lane];
        acc0 = __builtin_amdgcn_wmma_f32_16x16x32_bf16(
            false, a0[t], false, b, (short)0, acc0, false, false);
        acc1 = __builtin_amdgcn_wmma_f32_16x16x32_bf16(
            false, a1[t], false, b, (short)0, acc1, false, false);
      }
      const int   n   = c * 16 + mlo;
      const float w2n = w2[n];
#pragma unroll
      for (int r = 0; r < 8; ++r) {
        float s0 = w2n - 2.0f * acc0[r];
        float s1 = w2n - 2.0f * acc1[r];
        if (s0 < bestVal0[r]) { bestVal0[r] = s0; bestIdx0[r] = n; }
        if (s1 < bestVal1[r]) { bestVal1[r] = s1; bestIdx1[r] = n; }
      }
    }

    wait_asynccnt0();   // own async stage of next chunk complete
    __syncthreads();    // all waves done staging + computing this chunk
  }

  // Cross-lane argmin within each 16-lane half (half0 -> M=r, half1 -> M=8+r).
#pragma unroll
  for (int off = 8; off >= 1; off >>= 1) {
#pragma unroll
    for (int r = 0; r < 8; ++r) {
      float ov; int oi;
      ov = __shfl_xor(bestVal0[r], off, 32); oi = __shfl_xor(bestIdx0[r], off, 32);
      if (ov < bestVal0[r] || (ov == bestVal0[r] && oi < bestIdx0[r])) {
        bestVal0[r] = ov; bestIdx0[r] = oi;
      }
      ov = __shfl_xor(bestVal1[r], off, 32); oi = __shfl_xor(bestIdx1[r], off, 32);
      if (ov < bestVal1[r] || (ov == bestVal1[r] && oi < bestIdx1[r])) {
        bestVal1[r] = ov; bestIdx1[r] = oi;
      }
    }
  }
  if (mlo == 0) {
#pragma unroll
    for (int r = 0; r < 8; ++r) {
      s_idx[wave * 32 + hi * 8 + r]      = bestIdx0[r];
      s_idx[wave * 32 + 16 + hi * 8 + r] = bestIdx1[r];
    }
  }
  __syncthreads();

  // Gather fp32 codebook rows: 32 rows x 256 floats per wave.
  for (int i = 0; i < 32; ++i) {
    const int idx = s_idx[wave * 32 + i];
    const float4* src = (const float4*)(W + (size_t)idx * VEC_DIM);
    float4* dst = (float4*)(q + (mBase0 + (size_t)i) * VEC_DIM);
    dst[lane]      = src[lane];
    dst[lane + 32] = src[lane + 32];
  }
}

// ---------------------------------------------------------------------------
// Codebook min pairwise distance (tiny): block min-reduce + atomicMin on bits.
// ---------------------------------------------------------------------------
__global__ void init_min(unsigned* minBits) { *minBits = 0x7F800000u; }

__global__ void codebook_mindist(const float* __restrict__ W,
                                 unsigned* __restrict__ minBits) {
  __shared__ float wi[VEC_DIM];
  __shared__ float red[256];
  const int i   = blockIdx.x;      // row 0..1023
  const int tid = threadIdx.x;     // 256 threads
  wi[tid] = W[(size_t)i * VEC_DIM + tid];
  __syncthreads();

  float best = __builtin_inff();
  for (int j = tid; j < NUM_Q; j += 256) {
    if (j == i) continue;
    const float4* row = (const float4*)(W + (size_t)j * VEC_DIM);
    float s = 0.f;
#pragma unroll 4
    for (int u = 0; u < VEC_DIM / 4; ++u) {
      float4 v = row[u];
      float d0 = wi[u * 4 + 0] - v.x, d1 = wi[u * 4 + 1] - v.y;
      float d2 = wi[u * 4 + 2] - v.z, d3 = wi[u * 4 + 3] - v.w;
      s += d0 * d0 + d1 * d1 + d2 * d2 + d3 * d3;
    }
    best = fminf(best, s);
  }
  red[tid] = best;
  __syncthreads();
  for (int sft = 128; sft > 0; sft >>= 1) {
    if (tid < sft) red[tid] = fminf(red[tid], red[tid + sft]);
    __syncthreads();
  }
  if (tid == 0) atomicMin(minBits, __float_as_uint(red[0]));
}

__global__ void finish_min(const unsigned* __restrict__ minBits,
                           float* __restrict__ out) {
  out[0] = sqrtf(__uint_as_float(*minBits));
}

// ---------------------------------------------------------------------------
extern "C" void kernel_launch(void* const* d_in, const int* in_sizes, int n_in,
                              void* d_out, int out_size, void* d_ws, size_t ws_size,
                              hipStream_t stream) {
  const float* X = (const float*)d_in[0];   // [32*4096, 256] fp32
  const float* W = (const float*)d_in[1];   // [1024, 256]   fp32
  float* out = (float*)d_out;               // q (N*256 floats) ++ min_dist (1)

  char* ws = (char*)d_ws;
  __bf16*   Wf      = (__bf16*)(ws);                       // 512 KB fragments
  float*    w2      = (float*)(ws + 512 * 1024);           // 4 KB
  unsigned* minBits = (unsigned*)(ws + 512 * 1024 + 4096); // 4 B

  prep_wfrag<<<64, 256, 0, stream>>>(W, Wf);
  prep_w2<<<4, 256, 0, stream>>>(W, w2);
  init_min<<<1, 1, 0, stream>>>(minBits);

  const size_t lds_bytes = 2 * CHUNK_BYTES + 8 * 32 * sizeof(int); // 66560
  vq_assign_gather<<<512, 256, lds_bytes, stream>>>(X, W, Wf, w2, out);

  codebook_mindist<<<1024, 256, 0, stream>>>(W, minBits);
  finish_min<<<1, 1, 0, stream>>>(minBits, out + (size_t)N_ROWS * VEC_DIM);
}
